// LlamaAttention_78752520339586
// MI455X (gfx1250) — matrix operand order repaired and checked by reading, not compiled, not run on hardware
//
#include <hip/hip_runtime.h>
#include <hip/hip_bf16.h>

// ---------------- problem constants ----------------
static constexpr int T_   = 2048;
static constexpr int H_   = 4096;
static constexpr int NH_  = 32;
static constexpr int NKV_ = 8;
static constexpr int HD_  = 128;
static constexpr int QKV_N = NH_ * HD_ + 2 * NKV_ * HD_;   // 6144
static constexpr float ROPE_THETA_ = 500000.0f;

// ---------------- WMMA vector types ----------------
typedef __attribute__((ext_vector_type(16))) __bf16 v16bf;
typedef __attribute__((ext_vector_type(8)))  __bf16 v8bf;
typedef __attribute__((ext_vector_type(8)))  float  v8f;

__device__ __forceinline__ v16bf frag_combine(v8bf lo, v8bf hi) {
    v16bf r;
#pragma unroll
    for (int i = 0; i < 8; ++i) { r[i] = lo[i]; r[i + 8] = hi[i]; }
    return r;
}

__device__ __forceinline__ v8f wmma_bf16(v16bf a, v16bf b, v8f c) {
    // D = A(16x32) * B(32x16) + C(16x16), f32 accumulate
    return __builtin_amdgcn_wmma_f32_16x16x32_bf16(
        /*neg_a=*/false, a, /*neg_b=*/false, b,
        /*c_mod=*/(short)0, c, /*reuse_a=*/false, /*reuse_b=*/false);
}

// ---- CDNA5 async global->LDS copy (ASYNCcnt-tracked, no VGPR data path) ----
// IOFFSET is applied to BOTH the LDS and global addresses
// so one base covers a 64-byte row with immediate offsets.
__device__ __forceinline__ void async_row64B_to_lds(void* lds_dst, const void* gsrc) {
    unsigned l = (unsigned)(uintptr_t)lds_dst;            // low 32 bits = LDS offset
    unsigned long long g = (unsigned long long)(uintptr_t)gsrc;
    asm volatile(
        "global_load_async_to_lds_b128 %0, %1, off\n\t"
        "global_load_async_to_lds_b128 %0, %1, off offset:16\n\t"
        "global_load_async_to_lds_b128 %0, %1, off offset:32\n\t"
        "global_load_async_to_lds_b128 %0, %1, off offset:48"
        :: "v"(l), "v"(g) : "memory");
}
__device__ __forceinline__ void wait_async_all() {
    asm volatile("s_wait_asynccnt 0" ::: "memory");
}

// ---------------- f32 -> bf16 (x4 vectorized, no transpose) ----------------
__global__ void cvt_f32_bf16(const float* __restrict__ in, __bf16* __restrict__ out, int n4) {
    int i = blockIdx.x * blockDim.x + threadIdx.x;
    if (i < n4) {
        float4 v = ((const float4*)in)[i];
        alignas(8) __bf16 o[4] = {(__bf16)v.x, (__bf16)v.y, (__bf16)v.z, (__bf16)v.w};
        *(ushort4*)(out + (size_t)4 * i) = *(const ushort4*)o;
    }
}

// ---------------- f32 -> bf16 transposed: out[n][k] = (bf16) in[k][n] ----------------
__global__ __launch_bounds__(256)
void cvt_transpose_bf16(const float* __restrict__ in, __bf16* __restrict__ out,
                        int K /*rows of in*/, int N /*cols of in*/) {
    __shared__ __bf16 tile[32][33];
    const int tx = threadIdx.x & 31;
    const int ty = threadIdx.x >> 5;     // 0..7
    const int k0 = blockIdx.y * 32;
    const int n0 = blockIdx.x * 32;
#pragma unroll
    for (int i = 0; i < 4; ++i) {
        int k = k0 + ty + i * 8;
        tile[tx][ty + i * 8] = (__bf16)in[(size_t)k * N + n0 + tx];
    }
    __syncthreads();
#pragma unroll
    for (int i = 0; i < 4; ++i) {
        int n = n0 + ty + i * 8;
        out[(size_t)n * K + k0 + tx] = tile[ty + i * 8][tx];
    }
}

// ---------------- GEMM: C(f32) = A(bf16, MxK) * BT(bf16, NxK) ----------------
// Block tile 128x128, K-step 64, double-buffered LDS staged with async b128
// copies overlapped against the 16 WMMAs of the other buffer. 8 waves (2M x 4N).
// Requires K % 128 == 0.
__global__ __launch_bounds__(256)
void gemm_bf16(const __bf16* __restrict__ A, const __bf16* __restrict__ BT,
               float* __restrict__ C, int M, int N, int K) {
    __shared__ __bf16 a_t[2][128][64];   // [buf][m][k]  32 KB
    __shared__ __bf16 b_t[2][128][64];   // [buf][n][k]  32 KB

    const int tid  = threadIdx.x;
    const int lane = tid & 31;
    const int wid  = tid >> 5;
    const int wm   = wid >> 2;          // 0..1
    const int wn   = wid & 3;           // 0..3
    const int m0   = blockIdx.y * 128;
    const int n0   = blockIdx.x * 128;
    const int mlane = lane & 15;
    const int khi   = lane >> 4;        // 0/1

    v8f acc[4][2];
#pragma unroll
    for (int i = 0; i < 4; ++i)
#pragma unroll
        for (int j = 0; j < 2; ++j) acc[i][j] = (v8f){0,0,0,0,0,0,0,0};

    // staging: 2 threads per row, one 64-byte row chunk each, both tiles
    const int ar = tid >> 1;
    const int ak = (tid & 1) * 32;
    const __bf16* agBase = A  + (size_t)(m0 + ar) * K + ak;
    const __bf16* bgBase = BT + (size_t)(n0 + ar) * K + ak;

    auto stage = [&](int buf, int kb) {
        async_row64B_to_lds(&a_t[buf][ar][ak], agBase + kb);
        async_row64B_to_lds(&b_t[buf][ar][ak], bgBase + kb);
    };
    auto compute = [&](const __bf16 (*at)[64], const __bf16 (*bt)[64]) {
#pragma unroll
        for (int kc = 0; kc < 2; ++kc) {
            v16bf afr[4], bfr[2];
#pragma unroll
            for (int i = 0; i < 4; ++i) {
                // A 16x32 lane layout: m=lane&15; vgpr0-3: k=(lane>>4)*8+0..7, vgpr4-7: +16
                const __bf16* p = &at[wm * 64 + i * 16 + mlane][kc * 32 + khi * 8];
                afr[i] = frag_combine(*(const v8bf*)p, *(const v8bf*)(p + 16));
            }
#pragma unroll
            for (int j = 0; j < 2; ++j) {
                // B 32x16 lane layout: n=lane&15; vgpr0-3: k=(lane>>4)*8+0..7, vgpr4-7: +16
                const __bf16* p = &bt[wn * 32 + j * 16 + mlane][kc * 32 + khi * 8];
                bfr[j] = frag_combine(*(const v8bf*)p, *(const v8bf*)(p + 16));
            }
#pragma unroll
            for (int i = 0; i < 4; ++i)
#pragma unroll
                for (int j = 0; j < 2; ++j)
                    acc[i][j] = wmma_bf16(afr[i], bfr[j], acc[i][j]);
        }
    };

    stage(0, 0);
    for (int kb = 0; kb < K; kb += 128) {
        // buffer 0 half-step
        wait_async_all();        // own copies for buf0 done
        __syncthreads();         // everyone past reads of buf1, buf0 globally ready
        stage(1, kb + 64);       // overlap next stage with compute (K%128==0 -> always valid)
        compute(a_t[0], b_t[0]);
        // buffer 1 half-step
        wait_async_all();
        __syncthreads();
        if (kb + 128 < K) stage(0, kb + 128);
        compute(a_t[1], b_t[1]);
    }

    // C layout: lane l, vgpr r -> m = (l>>4)*8 + r, n = l&15
#pragma unroll
    for (int i = 0; i < 4; ++i)
#pragma unroll
        for (int j = 0; j < 2; ++j)
#pragma unroll
            for (int r = 0; r < 8; ++r) {
                int m = m0 + wm * 64 + i * 16 + khi * 8 + r;
                int n = n0 + wn * 32 + j * 16 + mlane;
                C[(size_t)m * N + n] = acc[i][j][r];
            }
}

// ---------------- RoPE + repack ----------------
// Qb[t][h][d] bf16; Kb[kvh][s][d] bf16; Vt[kvh][d][s] bf16
__global__ __launch_bounds__(256)
void rope_pack(const float* __restrict__ qkv, const int* __restrict__ positions,
               __bf16* __restrict__ Qb, __bf16* __restrict__ Kb, __bf16* __restrict__ Vt) {
    const int t = blockIdx.x;
    const float pos = (float)positions[t];
    const float* row = qkv + (size_t)t * QKV_N;

    for (int idx = threadIdx.x; idx < NH_ * 64; idx += blockDim.x) {
        int h = idx >> 6, d = idx & 63;
        float inv = __powf(ROPE_THETA_, -(float)d * (1.0f / 64.0f));
        float s, c;
        __sincosf(pos * inv, &s, &c);
        float x1 = row[h * HD_ + d];
        float x2 = row[h * HD_ + d + 64];
        size_t base = (size_t)t * (NH_ * HD_) + h * HD_;
        Qb[base + d]      = (__bf16)(x1 * c - x2 * s);
        Qb[base + d + 64] = (__bf16)(x2 * c + x1 * s);
    }
    for (int idx = threadIdx.x; idx < NKV_ * 64; idx += blockDim.x) {
        int h = idx >> 6, d = idx & 63;
        float inv = __powf(ROPE_THETA_, -(float)d * (1.0f / 64.0f));
        float s, c;
        __sincosf(pos * inv, &s, &c);
        float x1 = row[NH_ * HD_ + h * HD_ + d];
        float x2 = row[NH_ * HD_ + h * HD_ + d + 64];
        size_t base = ((size_t)h * T_ + t) * HD_;
        Kb[base + d]      = (__bf16)(x1 * c - x2 * s);
        Kb[base + d + 64] = (__bf16)(x2 * c + x1 * s);
    }
    for (int idx = threadIdx.x; idx < NKV_ * HD_; idx += blockDim.x) {
        int h = idx >> 7, d = idx & 127;
        Vt[((size_t)h * HD_ + d) * T_ + t] =
            (__bf16)row[NH_ * HD_ + NKV_ * HD_ + h * HD_ + d];
    }
}

// ---------------- causal flash attention ----------------
// grid = (NH, T/128). 8 waves; each wave owns 16 query rows, full d=128.
__global__ __launch_bounds__(256)
void attn_fwd(const __bf16* __restrict__ Qb, const __bf16* __restrict__ Kb,
              const __bf16* __restrict__ Vt, __bf16* __restrict__ Ob) {
    __shared__ __bf16 p_lds[8][16][32];   // per-wave P staging (C->A layout swap)

    const int h    = blockIdx.x;
    const int kvh  = h >> 2;              // group = NH/NKV = 4
    const int lane = threadIdx.x & 31;
    const int wid  = threadIdx.x >> 5;
    const int qrow0 = blockIdx.y * 128 + wid * 16;
    const int mlane = lane & 15;
    const int khi   = lane >> 4;
    const float scale = 0.08838834764831845f;   // 128^-0.5

    v16bf qfr[4];
    {
        const __bf16* qp = Qb + ((size_t)(qrow0 + mlane) * NH_ + h) * HD_;
#pragma unroll
        for (int kc = 0; kc < 4; ++kc) {
            const __bf16* p = qp + kc * 32 + khi * 8;
            qfr[kc] = frag_combine(*(const v8bf*)p, *(const v8bf*)(p + 16));
        }
    }

    v8f acc[8];
#pragma unroll
    for (int j = 0; j < 8; ++j) acc[j] = (v8f){0,0,0,0,0,0,0,0};
    float mrow[8], lrow[8];
#pragma unroll
    for (int r = 0; r < 8; ++r) { mrow[r] = -1e30f; lrow[r] = 0.0f; }

    const int send = qrow0 + 16;   // keys needed: s <= qrow0+15
    for (int sb = 0; sb < send; sb += 32) {
        v8f sc[2];
#pragma unroll
        for (int ns = 0; ns < 2; ++ns) {
            v8f s = (v8f){0,0,0,0,0,0,0,0};
#pragma unroll
            for (int kc = 0; kc < 4; ++kc) {
                const __bf16* kp = Kb + ((size_t)kvh * T_ + sb + ns * 16 + mlane) * HD_
                                      + kc * 32 + khi * 8;
                v16bf kfr = frag_combine(*(const v8bf*)kp, *(const v8bf*)(kp + 16));
                s = wmma_bf16(qfr[kc], kfr, s);
            }
            sc[ns] = s;
        }
#pragma unroll
        for (int ns = 0; ns < 2; ++ns)
#pragma unroll
            for (int r = 0; r < 8; ++r) {
                int q = qrow0 + khi * 8 + r;
                int sg = sb + ns * 16 + mlane;
                float v = sc[ns][r] * scale;
                sc[ns][r] = (sg <= q) ? v : -1e30f;
            }
        float pr0[8], pr1[8];
#pragma unroll
        for (int r = 0; r < 8; ++r) {
            float mx = fmaxf(sc[0][r], sc[1][r]);
            mx = fmaxf(mx, __shfl_xor(mx, 1, 32));
            mx = fmaxf(mx, __shfl_xor(mx, 2, 32));
            mx = fmaxf(mx, __shfl_xor(mx, 4, 32));
            mx = fmaxf(mx, __shfl_xor(mx, 8, 32));
            float newm = fmaxf(mrow[r], mx);
            float corr = __expf(mrow[r] - newm);
            float p0 = __expf(sc[0][r] - newm);
            float p1 = __expf(sc[1][r] - newm);
            float rs = p0 + p1;
            rs += __shfl_xor(rs, 1, 32);
            rs += __shfl_xor(rs, 2, 32);
            rs += __shfl_xor(rs, 4, 32);
            rs += __shfl_xor(rs, 8, 32);
            lrow[r] = lrow[r] * corr + rs;
            mrow[r] = newm;
            pr0[r] = p0; pr1[r] = p1;
#pragma unroll
            for (int j = 0; j < 8; ++j) acc[j][r] *= corr;
        }
        __bf16* pb = &p_lds[wid][0][0];
#pragma unroll
        for (int r = 0; r < 8; ++r) {
            pb[(khi * 8 + r) * 32 + mlane]      = (__bf16)pr0[r];
            pb[(khi * 8 + r) * 32 + 16 + mlane] = (__bf16)pr1[r];
        }
        asm volatile("s_wait_dscnt 0" ::: "memory");
        const __bf16* pf = pb + mlane * 32 + khi * 8;
        v16bf pfr = frag_combine(*(const v8bf*)pf, *(const v8bf*)(pf + 16));
#pragma unroll
        for (int j = 0; j < 8; ++j) {
            const __bf16* vp = Vt + ((size_t)kvh * HD_ + j * 16 + mlane) * T_
                                  + sb + khi * 8;
            v16bf vfr = frag_combine(*(const v8bf*)vp, *(const v8bf*)(vp + 16));
            acc[j] = wmma_bf16(pfr, vfr, acc[j]);
        }
    }

#pragma unroll
    for (int r = 0; r < 8; ++r) {
        float inv = 1.0f / lrow[r];
        int t = qrow0 + khi * 8 + r;
#pragma unroll
        for (int j = 0; j < 8; ++j)
            Ob[((size_t)t * NH_ + h) * HD_ + j * 16 + mlane] = (__bf16)(acc[j][r] * inv);
    }
}

// ---------------- launcher ----------------
extern "C" void kernel_launch(void* const* d_in, const int* in_sizes, int n_in,
                              void* d_out, int out_size, void* d_ws, size_t ws_size,
                              hipStream_t stream) {
    const int*   positions = (const int*)d_in[0];
    const float* hidden    = (const float*)d_in[1];
    const float* w_qkv     = (const float*)d_in[2];
    const float* w_o       = (const float*)d_in[3];
    float* out = (float*)d_out;

    // Workspace (112 MB), regions reused across serialized stages:
    //  A [ 0, 16MB): hs_bf (gemm1 A)            -> Ab (attn out, gemm2 A)
    //  B [16, 64MB): wqkvT (gemm1 B, NxK)       -> Qb(16) + Kb(4) + Vt(4)
    //  C [64,112MB): qkv f32 (gemm1 C, rope in) -> woT (gemm2 B, NxK)
    char* ws = (char*)d_ws;
    const size_t MB = (size_t)1 << 20;
    __bf16* hs_bf = (__bf16*)(ws);
    __bf16* Ab    = (__bf16*)(ws);
    __bf16* wqkvT = (__bf16*)(ws + 16 * MB);
    __bf16* Qb    = (__bf16*)(ws + 16 * MB);
    __bf16* Kb    = (__bf16*)(ws + 32 * MB);
    __bf16* Vt    = (__bf16*)(ws + 36 * MB);
    float*  qkv   = (float*) (ws + 64 * MB);
    __bf16* woT   = (__bf16*)(ws + 64 * MB);

    // 1) activations -> bf16 (row-major, A operand)
    {
        int n4 = T_ * H_ / 4;
        cvt_f32_bf16<<<(n4 + 255) / 256, 256, 0, stream>>>(hidden, hs_bf, n4);
    }
    // 2) w_qkv -> bf16 transposed [N][K]
    cvt_transpose_bf16<<<dim3(QKV_N / 32, H_ / 32), 256, 0, stream>>>(
        w_qkv, wqkvT, H_, QKV_N);
    // 3) qkv projection
    gemm_bf16<<<dim3(QKV_N / 128, T_ / 128), 256, 0, stream>>>(
        hs_bf, wqkvT, qkv, T_, QKV_N, H_);
    // 4) rope + repack (consumes qkv and wqkvT regions)
    rope_pack<<<T_, 256, 0, stream>>>(qkv, positions, Qb, Kb, Vt);
    // 5) w_o -> bf16 transposed [N][K] (qkv region now dead)
    cvt_transpose_bf16<<<dim3(H_ / 32, H_ / 32), 256, 0, stream>>>(
        w_o, woT, H_, H_);
    // 6) attention (hs_bf region now dead)
    attn_fwd<<<dim3(NH_, T_ / 128), 256, 0, stream>>>(Qb, Kb, Vt, Ab);
    // 7) output projection
    gemm_bf16<<<dim3(H_ / 128, T_ / 128), 256, 0, stream>>>(
        Ab, woT, out, T_, H_, H_);
}
